// Stream1Encoder_62508954026087
// MI455X (gfx1250) — compile-verified, matrix operand order checked by reading, not run on hardware
//
#include <hip/hip_runtime.h>
#include <hip/hip_bf16.h>

// ---------------- problem constants (match reference) ----------------
#define Bc      4
#define Lc      1024
#define IN_DIM  64
#define D_MODEL 512
#define NL      3
#define D_INNER 1024
#define STATE   16
#define DT_RANK 32
#define KCONV   4
#define BL      (Bc * Lc)               // 4096 activation rows
#define PROJ_DIM (DT_RANK + 2 * STATE)  // 64

// ---------------- WMMA types ----------------
typedef __attribute__((ext_vector_type(16))) __bf16 v16bf;
typedef __attribute__((ext_vector_type(8)))  float  v8f;

// f32 -> bf16 round-to-nearest-even via bit ops
static __device__ __forceinline__ __bf16 f2bf(float f) {
    unsigned int u = __builtin_bit_cast(unsigned int, f);
    unsigned int r = (u + 0x7FFFu + ((u >> 16) & 1u)) >> 16;
    unsigned short s = (unsigned short)r;
    return __builtin_bit_cast(__bf16, s);
}

static __device__ __forceinline__ float sigmoid_f(float x) {
    return 1.0f / (1.0f + expf(-x));
}
static __device__ __forceinline__ float softplus_f(float x) {
    return (x > 20.0f) ? x : log1pf(expf(x));
}

// ---------------------------------------------------------------------
// Fragment load: for row-major bf16 with contiguous K, the wave32 WMMA
// A/B fragment (ISA 7.12.2) is two 16-byte loads per lane:
//   lane l: base = row(l&15)*ld + (l>>4)*8 ; elements [base, base+8) -> v0..3
//           and   [base+16, base+24)                                -> v4..7
// ---------------------------------------------------------------------
struct Frag128 { uint4 a, b; };
static __device__ __forceinline__ v16bf ldfrag(const __bf16* p) {
    Frag128 u;
    u.a = *(const uint4*)(p);
    u.b = *(const uint4*)(p + 16);
    return __builtin_bit_cast(v16bf, u);
}

// ---------------------------------------------------------------------
// GEMM: out[M x N] = A[M x Kdim](bf16) @ W[N x Kdim](bf16)^T (+bias)(+softplus)
// One wave per 32x32 output macro-tile: 2 A-frags x 2 B-frags ->
// 4 x v_wmma_f32_16x16x32_bf16 per K-step, f32 accumulation.
//   grid.x = N/32, grid.y = M/32, block = 32
// mode: 0 = +bias ; 1 = softplus(x + bias).  outbf: optional bf16 copy.
// ---------------------------------------------------------------------
__global__ void gemm_bf16_wmma(const __bf16* __restrict__ A, int lda,
                               const __bf16* __restrict__ W, int ldw,
                               const float* __restrict__ bias,
                               float* __restrict__ out, int ldo,
                               __bf16* __restrict__ outbf,
                               int Kdim, int mode) {
    const int lane  = threadIdx.x;        // 0..31
    const int half  = lane >> 4;
    const int idx15 = lane & 15;
    const long mtile = (long)blockIdx.y * 32;
    const long ntile = (long)blockIdx.x * 32;

    const __bf16* a0 = A + (mtile + idx15) * (long)lda + half * 8;
    const __bf16* a1 = a0 + 16 * (long)lda;
    const __bf16* w0 = W + (ntile + idx15) * (long)ldw + half * 8;
    const __bf16* w1 = w0 + 16 * (long)ldw;

    v8f acc00 = {}, acc01 = {}, acc10 = {}, acc11 = {};
    for (int k0 = 0; k0 < Kdim; k0 += 32) {
        const v16bf fa0 = ldfrag(a0 + k0);
        const v16bf fa1 = ldfrag(a1 + k0);
        const v16bf fb0 = ldfrag(w0 + k0);
        const v16bf fb1 = ldfrag(w1 + k0);
        acc00 = __builtin_amdgcn_wmma_f32_16x16x32_bf16(false, fa0, false, fb0, (short)0, acc00, false, false);
        acc01 = __builtin_amdgcn_wmma_f32_16x16x32_bf16(false, fa0, false, fb1, (short)0, acc01, false, false);
        acc10 = __builtin_amdgcn_wmma_f32_16x16x32_bf16(false, fa1, false, fb0, (short)0, acc10, false, false);
        acc11 = __builtin_amdgcn_wmma_f32_16x16x32_bf16(false, fa1, false, fb1, (short)0, acc11, false, false);
    }

    const long n0 = ntile + idx15;
    const long n1 = n0 + 16;
    const float bv0 = bias ? bias[n0] : 0.0f;
    const float bv1 = bias ? bias[n1] : 0.0f;
#pragma unroll
    for (int r = 0; r < 8; ++r) {
        const long m0 = mtile + r + (half << 3);  // C/D layout: VGPR r -> M=r(+8)
        const long m1 = m0 + 16;
        float v00 = acc00[r] + bv0, v01 = acc01[r] + bv1;
        float v10 = acc10[r] + bv0, v11 = acc11[r] + bv1;
        if (mode == 1) {
            v00 = softplus_f(v00); v01 = softplus_f(v01);
            v10 = softplus_f(v10); v11 = softplus_f(v11);
        }
        out[m0 * (long)ldo + n0] = v00;
        out[m0 * (long)ldo + n1] = v01;
        out[m1 * (long)ldo + n0] = v10;
        out[m1 * (long)ldo + n1] = v11;
        if (outbf) {
            outbf[m0 * (long)ldo + n0] = f2bf(v00);
            outbf[m0 * (long)ldo + n1] = f2bf(v01);
            outbf[m1 * (long)ldo + n0] = f2bf(v10);
            outbf[m1 * (long)ldo + n1] = f2bf(v11);
        }
    }
}

// one-time f32 -> bf16 tensor conversion (weights / input)
__global__ void f32_to_bf16_kernel(const float* __restrict__ src,
                                   __bf16* __restrict__ dst, long n) {
    const long t = (long)blockIdx.x * blockDim.x + threadIdx.x;
    if (t < n) dst[t] = f2bf(src[t]);
}

// ---------------------------------------------------------------------
// Depthwise causal conv (K=4) + SiLU; emits f32 (for scan) and bf16 (for GEMM).
// ---------------------------------------------------------------------
__global__ void conv_silu_kernel(const float* __restrict__ xz,
                                 const float* __restrict__ cw,
                                 const float* __restrict__ cb,
                                 float* __restrict__ xc,
                                 __bf16* __restrict__ xcbf) {
    const long t = (long)blockIdx.x * blockDim.x + threadIdx.x;
    if (t >= (long)BL * D_INNER) return;
    const int  d   = (int)(t % D_INNER);
    const long row = t / D_INNER;              // b*L + l
    const int  l   = (int)(row % Lc);
    float acc = cb[d];
#pragma unroll
    for (int k = 0; k < KCONV; ++k) {
        const int dl = k - (KCONV - 1);        // -3..0
        if (l + dl >= 0)
            acc += cw[d * KCONV + k] * xz[(row + dl) * (long)(2 * D_INNER) + d];
    }
    const float v = acc * sigmoid_f(acc);
    xc[t]   = v;
    xcbf[t] = f2bf(v);
}

// ---------------------------------------------------------------------
// Selective scan: one thread per (b,d), 16 f32 states in registers,
// sequential over L.  proj rows: [0:32)=dt_low, [32:48)=B, [48:64)=C.
// ---------------------------------------------------------------------
__global__ void scan_kernel(const float* __restrict__ dtb,
                            const float* __restrict__ proj,
                            const float* __restrict__ xc,
                            const float* __restrict__ A_log,
                            float* __restrict__ yb) {
    const int t = blockIdx.x * blockDim.x + threadIdx.x;
    if (t >= Bc * D_INNER) return;
    const int b = t / D_INNER;
    const int d = t % D_INNER;

    float Aloc[STATE], h[STATE];
#pragma unroll
    for (int n = 0; n < STATE; ++n) {
        Aloc[n] = -expf(A_log[d * STATE + n]);
        h[n] = 0.0f;
    }

    for (int l = 0; l < Lc; ++l) {
        const long row = (long)b * Lc + l;
        if (l + 8 < Lc)  // prefetch upcoming B/C rows (global_prefetch_b8)
            __builtin_prefetch(proj + (row + 8) * PROJ_DIM, 0, 1);
        const float dt = dtb[row * D_INNER + d];
        const float xv = xc [row * D_INNER + d];
        const float dx = dt * xv;
        const float* pr = proj + row * PROJ_DIM;
        float y = 0.0f;
#pragma unroll
        for (int n = 0; n < STATE; ++n) {
            h[n] = h[n] * expf(dt * Aloc[n]) + dx * pr[DT_RANK + n];
            y += h[n] * pr[DT_RANK + STATE + n];
        }
        yb[row * D_INNER + d] = y;
    }
}

// y = (ys + xc * D[d]) * silu(z) -> bf16 (consumed only by out_proj GEMM)
__global__ void gate_kernel(const float* __restrict__ xz,
                            const float* __restrict__ xc,
                            const float* __restrict__ Dp,
                            const float* __restrict__ yb,
                            __bf16* __restrict__ ybf) {
    const long t = (long)blockIdx.x * blockDim.x + threadIdx.x;
    if (t >= (long)BL * D_INNER) return;
    const int  d   = (int)(t % D_INNER);
    const long row = t / D_INNER;
    const float z = xz[row * (long)(2 * D_INNER) + D_INNER + d];
    const float y = yb[t] + xc[t] * Dp[d];
    ybf[t] = f2bf(y * z * sigmoid_f(z));
}

// h = LayerNorm(tmp + h); emits f32 residual stream + bf16 for next in_proj.
__global__ void add_ln_kernel(const float* __restrict__ tmp,
                              float* __restrict__ hbuf,
                              __bf16* __restrict__ hbf,
                              const float* __restrict__ lw,
                              const float* __restrict__ lb) {
    const long row  = blockIdx.x;
    const int  lane = threadIdx.x;           // 0..31, one wave per row
    float v[D_MODEL / 32];
    float s = 0.0f;
#pragma unroll
    for (int i = 0; i < D_MODEL / 32; ++i) {
        const int c = lane + i * 32;
        v[i] = tmp[row * D_MODEL + c] + hbuf[row * D_MODEL + c];
        s += v[i];
    }
#pragma unroll
    for (int m = 16; m >= 1; m >>= 1) s += __shfl_xor(s, m, 32);
    const float mu = s * (1.0f / D_MODEL);
    float s2 = 0.0f;
#pragma unroll
    for (int i = 0; i < D_MODEL / 32; ++i) {
        const float dvi = v[i] - mu;
        s2 += dvi * dvi;
    }
#pragma unroll
    for (int m = 16; m >= 1; m >>= 1) s2 += __shfl_xor(s2, m, 32);
    const float rstd = rsqrtf(s2 * (1.0f / D_MODEL) + 1e-5f);
#pragma unroll
    for (int i = 0; i < D_MODEL / 32; ++i) {
        const int c = lane + i * 32;
        const float o = (v[i] - mu) * rstd * lw[c] + lb[c];
        hbuf[row * D_MODEL + c] = o;
        hbf [row * D_MODEL + c] = f2bf(o);
    }
}

// out[b, :] = h[b, L-1, :]
__global__ void extract_kernel(const float* __restrict__ hbuf,
                               float* __restrict__ out) {
    const int t = blockIdx.x * blockDim.x + threadIdx.x;
    if (t >= Bc * D_MODEL) return;
    const int b = t / D_MODEL;
    const int c = t % D_MODEL;
    out[t] = hbuf[((long)b * Lc + (Lc - 1)) * D_MODEL + c];
}

// ---------------------------------------------------------------------
extern "C" void kernel_launch(void* const* d_in, const int* in_sizes, int n_in,
                              void* d_out, int out_size, void* d_ws, size_t ws_size,
                              hipStream_t stream) {
    (void)in_sizes; (void)n_in; (void)out_size; (void)ws_size;

    const float* x          = (const float*)d_in[0];   // (4,1024,64)
    const float* input_w    = (const float*)d_in[1];   // (512,64)
    const float* input_b    = (const float*)d_in[2];   // (512)
    const float* in_proj_w  = (const float*)d_in[3];   // (3,2048,512)
    const float* in_proj_b  = (const float*)d_in[4];   // (3,2048)
    const float* conv_w     = (const float*)d_in[5];   // (3,1024,1,4)
    const float* conv_b     = (const float*)d_in[6];   // (3,1024)
    const float* x_proj_w   = (const float*)d_in[7];   // (3,64,1024)
    const float* dt_proj_w  = (const float*)d_in[8];   // (3,1024,32)
    const float* dt_proj_b  = (const float*)d_in[9];   // (3,1024)
    const float* A_log      = (const float*)d_in[10];  // (3,1024,16)
    const float* Dp         = (const float*)d_in[11];  // (3,1024)
    const float* out_proj_w = (const float*)d_in[12];  // (3,512,1024)
    const float* out_proj_b = (const float*)d_in[13];  // (3,512)
    const float* ln_w       = (const float*)d_in[14];  // (3,512)
    const float* ln_b       = (const float*)d_in[15];  // (3,512)
    float* out = (float*)d_out;                        // (4,512)

    // ---- carve workspace ----
    char* base = (char*)d_ws;
    size_t off = 0;
    auto carve_f = [&](size_t elems) {
        char* p = base + off;
        off += (elems * 4 + 255) & ~(size_t)255;
        return (float*)p;
    };
    auto carve_b = [&](size_t elems) {
        char* p = base + off;
        off += (elems * 2 + 255) & ~(size_t)255;
        return (__bf16*)p;
    };
    // f32 activations
    float* hbuf = carve_f((size_t)BL * D_MODEL);      // residual stream
    float* xz   = carve_f((size_t)BL * 2 * D_INNER);  // in_proj out
    float* xc   = carve_f((size_t)BL * D_INNER);      // conv+silu out
    float* prj  = carve_f((size_t)BL * PROJ_DIM);     // x_proj out
    float* dtb  = carve_f((size_t)BL * D_INNER);      // softplus(dt)
    float* yb   = carve_f((size_t)BL * D_INNER);      // scan out
    float* tmp  = carve_f((size_t)BL * D_MODEL);      // out_proj out
    // bf16 activations
    __bf16* x_bf   = carve_b((size_t)BL * IN_DIM);
    __bf16* h_bf   = carve_b((size_t)BL * D_MODEL);
    __bf16* xc_bf  = carve_b((size_t)BL * D_INNER);
    __bf16* prj_bf = carve_b((size_t)BL * PROJ_DIM);
    __bf16* yb_bf  = carve_b((size_t)BL * D_INNER);
    // bf16 weights
    __bf16* input_w_bf    = carve_b((size_t)D_MODEL * IN_DIM);
    __bf16* in_proj_w_bf  = carve_b((size_t)NL * 2 * D_INNER * D_MODEL);
    __bf16* x_proj_w_bf   = carve_b((size_t)NL * PROJ_DIM * D_INNER);
    __bf16* dt_proj_w_bf  = carve_b((size_t)NL * D_INNER * DT_RANK);
    __bf16* out_proj_w_bf = carve_b((size_t)NL * D_MODEL * D_INNER);

    auto cvt = [&](const float* s, __bf16* d, long n) {
        f32_to_bf16_kernel<<<(n + 255) / 256, 256, 0, stream>>>(s, d, n);
    };
    cvt(x,          x_bf,          (long)BL * IN_DIM);
    cvt(input_w,    input_w_bf,    (long)D_MODEL * IN_DIM);
    cvt(in_proj_w,  in_proj_w_bf,  (long)NL * 2 * D_INNER * D_MODEL);
    cvt(x_proj_w,   x_proj_w_bf,   (long)NL * PROJ_DIM * D_INNER);
    cvt(dt_proj_w,  dt_proj_w_bf,  (long)NL * D_INNER * DT_RANK);
    cvt(out_proj_w, out_proj_w_bf, (long)NL * D_MODEL * D_INNER);

    const dim3 wave(32);

    // h = x @ input_w.T + input_b      (4096x64)x(512x64)^T
    gemm_bf16_wmma<<<dim3(D_MODEL / 32, BL / 32), wave, 0, stream>>>(
        x_bf, IN_DIM, input_w_bf, IN_DIM, input_b,
        hbuf, D_MODEL, h_bf, IN_DIM, 0);

    for (int i = 0; i < NL; ++i) {
        // xz = h @ in_proj_w[i].T + in_proj_b[i]
        gemm_bf16_wmma<<<dim3(2 * D_INNER / 32, BL / 32), wave, 0, stream>>>(
            h_bf, D_MODEL, in_proj_w_bf + (size_t)i * 2 * D_INNER * D_MODEL, D_MODEL,
            in_proj_b + (size_t)i * 2 * D_INNER, xz, 2 * D_INNER, nullptr, D_MODEL, 0);

        // xc = silu(causal_conv(xp))
        conv_silu_kernel<<<((long)BL * D_INNER + 255) / 256, 256, 0, stream>>>(
            xz, conv_w + (size_t)i * D_INNER * KCONV,
            conv_b + (size_t)i * D_INNER, xc, xc_bf);

        // proj = xc @ x_proj_w[i].T   (no bias)
        gemm_bf16_wmma<<<dim3(PROJ_DIM / 32, BL / 32), wave, 0, stream>>>(
            xc_bf, D_INNER, x_proj_w_bf + (size_t)i * PROJ_DIM * D_INNER, D_INNER,
            nullptr, prj, PROJ_DIM, prj_bf, D_INNER, 0);

        // dt = softplus(proj[:, :32] @ dt_proj_w[i].T + dt_proj_b[i])
        gemm_bf16_wmma<<<dim3(D_INNER / 32, BL / 32), wave, 0, stream>>>(
            prj_bf, PROJ_DIM, dt_proj_w_bf + (size_t)i * D_INNER * DT_RANK, DT_RANK,
            dt_proj_b + (size_t)i * D_INNER, dtb, D_INNER, nullptr, DT_RANK, 1);

        // selective scan
        scan_kernel<<<(Bc * D_INNER + 255) / 256, 256, 0, stream>>>(
            dtb, prj, xc, A_log + (size_t)i * D_INNER * STATE, yb);

        // y = (ys + xc*D) * silu(z)  -> bf16
        gate_kernel<<<((long)BL * D_INNER + 255) / 256, 256, 0, stream>>>(
            xz, xc, Dp + (size_t)i * D_INNER, yb, yb_bf);

        // tmp = y @ out_proj_w[i].T + out_proj_b[i]
        gemm_bf16_wmma<<<dim3(D_MODEL / 32, BL / 32), wave, 0, stream>>>(
            yb_bf, D_INNER, out_proj_w_bf + (size_t)i * D_MODEL * D_INNER, D_INNER,
            out_proj_b + (size_t)i * D_MODEL, tmp, D_MODEL, nullptr, D_INNER, 0);

        // h = LayerNorm(tmp + h)
        add_ln_kernel<<<BL, 32, 0, stream>>>(
            tmp, hbuf, h_bf, ln_w + (size_t)i * D_MODEL, ln_b + (size_t)i * D_MODEL);
    }

    extract_kernel<<<(Bc * D_MODEL + 255) / 256, 256, 0, stream>>>(hbuf, out);
}
